// BigBirdBlock_90572270338580
// MI455X (gfx1250) — matrix operand order repaired and checked by reading, not compiled, not run on hardware
//
#include <hip/hip_runtime.h>
#include <hip/hip_bf16.h>
#include <math.h>

// ---------------------------------------------------------------------------
// BigBird transformer block for MI455X (gfx1250, wave32, WMMA).
// All matmuls via v_wmma_f32_16x16x32_f16 (f16 in, f32 accum).
// Weights pre-transposed to [N][K] f16 so every LDS tile (A and B) is staged
// with global_load_async_to_lds_b128.
// ---------------------------------------------------------------------------

typedef __attribute__((ext_vector_type(16))) _Float16 v16h;
typedef __attribute__((ext_vector_type(8)))  _Float16 v8h;
typedef __attribute__((ext_vector_type(8)))  float    v8f;
typedef int v4i __attribute__((vector_size(16)));

#define FULL_SHUF(lo, hi) \
  __builtin_shufflevector(lo, hi, 0, 1, 2, 3, 4, 5, 6, 7, 8, 9, 10, 11, 12, 13, 14, 15)

#if defined(__AMDGCN__) && defined(__gfx1250__) && \
    __has_builtin(__builtin_amdgcn_global_load_async_to_lds_b128)
#define HAVE_ASYNC_LDS 1
#else
#define HAVE_ASYNC_LDS 0
#endif

// Copy 16 bytes (8 f16) global -> LDS. Async DMA path when available.
__device__ __forceinline__ void copy16_g2l(const _Float16* g, _Float16* l) {
#if HAVE_ASYNC_LDS
  typedef __attribute__((address_space(1))) v4i* gvp;  // global int32x4*
  typedef __attribute__((address_space(3))) v4i* lvp;  // LDS int32x4*
  __builtin_amdgcn_global_load_async_to_lds_b128((gvp)g, (lvp)l, 0, 0);
#else
  *(v8h*)l = *(const v8h*)g;
#endif
}

__device__ __forceinline__ void wait_async() {
#if HAVE_ASYNC_LDS
#if __has_builtin(__builtin_amdgcn_s_wait_asynccnt)
  __builtin_amdgcn_s_wait_asynccnt(0);
#else
  asm volatile("s_wait_asynccnt 0" ::: "memory");
#endif
#endif
}

__device__ __forceinline__ v8f wmma16(v16h a, v16h b, v8f c) {
#if defined(__AMDGCN__)
  return __builtin_amdgcn_wmma_f32_16x16x32_f16(
      /*neg_a=*/false, a, /*neg_b=*/false, b,
      /*c_mod=*/(short)0, c, /*reuse_a=*/false, /*reuse_b=*/false);
#else
  (void)a; (void)b;
  return c;
#endif
}

// ---------------------------------------------------------------------------
// Weight conversion + transpose: W[K][N] f32 -> WT[N][K] f16
// ---------------------------------------------------------------------------
__global__ void __launch_bounds__(256)
k_cvt_t(const float* __restrict__ src, _Float16* __restrict__ dst, int N, int K) {
  int i = blockIdx.x * 256 + threadIdx.x;   // i indexes WT flat: [n][k]
  if (i < N * K) {
    int n = i / K, k = i - n * K;
    dst[i] = (_Float16)src[(size_t)k * N + n];
  }
}

// ---------------------------------------------------------------------------
// BigBird attend-index table: [prev, self, next, 0, 63, 3 pseudo-random]
// ---------------------------------------------------------------------------
__global__ void k_build_idx(int* __restrict__ idx) {
  int i = threadIdx.x;
  if (i >= 64) return;
  int chosen[8];
  chosen[0] = (i + 63) & 63;
  chosen[1] = i;
  chosen[2] = (i + 1) & 63;
  chosen[3] = 0;
  chosen[4] = 63;
  unsigned state = 0x9E3779B9u ^ ((unsigned)i * 2654435761u);
  int cnt = 5;
  while (cnt < 8) {
    state = state * 1664525u + 1013904223u;
    int c = (int)(state >> 8) & 63;
    bool ok = true;
    for (int j = 0; j < cnt; ++j)
      if (chosen[j] == c) ok = false;
    if (ok) chosen[cnt++] = c;
  }
  for (int j = 0; j < 8; ++j) idx[i * 8 + j] = chosen[j];
}

// ---------------------------------------------------------------------------
// LayerNorm over D=768, one row per 256-thread block, f16 output
// ---------------------------------------------------------------------------
__global__ void __launch_bounds__(256)
k_layernorm(const float* __restrict__ x, const float* __restrict__ w,
            const float* __restrict__ b, _Float16* __restrict__ out) {
  const int row = blockIdx.x;
  const int t = threadIdx.x;
  const float* xr = x + (size_t)row * 768;
  float v0 = xr[t], v1 = xr[t + 256], v2 = xr[t + 512];
  __shared__ float red[8];

  float s = v0 + v1 + v2;
#pragma unroll
  for (int o = 16; o > 0; o >>= 1) s += __shfl_xor(s, o, 32);
  if ((t & 31) == 0) red[t >> 5] = s;
  __syncthreads();
  float mu = (red[0] + red[1] + red[2] + red[3] +
              red[4] + red[5] + red[6] + red[7]) * (1.0f / 768.0f);
  __syncthreads();

  float d0 = v0 - mu, d1 = v1 - mu, d2 = v2 - mu;
  float q = d0 * d0 + d1 * d1 + d2 * d2;
#pragma unroll
  for (int o = 16; o > 0; o >>= 1) q += __shfl_xor(q, o, 32);
  if ((t & 31) == 0) red[t >> 5] = q;
  __syncthreads();
  float var = (red[0] + red[1] + red[2] + red[3] +
               red[4] + red[5] + red[6] + red[7]) * (1.0f / 768.0f);
  float inv = rsqrtf(var + 1e-12f);

  _Float16* orow = out + (size_t)row * 768;
  orow[t]       = (_Float16)(d0 * inv * w[t]       + b[t]);
  orow[t + 256] = (_Float16)(d1 * inv * w[t + 256] + b[t + 256]);
  orow[t + 512] = (_Float16)(d2 * inv * w[t + 512] + b[t + 512]);
}

// ---------------------------------------------------------------------------
// WMMA GEMM: C[M,N] = act(A16[M,K] @ BT16[N,K]^T + bias [+ resid]),
// 64x64 C tile per block (8 waves x 16x32 strips), BK = 32.
// Both LDS tiles staged via async DMA (contiguous 16B segments).
// Optional extra f16 output transposed per-head: vt[b][h][d][s] (for V).
// ---------------------------------------------------------------------------
__global__ void __launch_bounds__(256)
k_gemm(const _Float16* __restrict__ A, const _Float16* __restrict__ BT,
       const float* __restrict__ bias, const float* __restrict__ resid,
       float* __restrict__ outF32, _Float16* __restrict__ outF16,
       _Float16* __restrict__ outVT,
       int M, int N, int K, int act) {
  __shared__ _Float16 As[64][40];   // [m][k], padded
  __shared__ _Float16 BsT[64][40];  // [n][k], padded

  const int bm = blockIdx.y * 64;
  const int bn = blockIdx.x * 64;
  const int tid = threadIdx.x;
  const int wave = tid >> 5, lane = tid & 31;
  const int wm = (wave & 3) * 16, wn = (wave >> 2) * 32;
  const int l15 = lane & 15, halfsel = lane >> 4;
  const int lr = tid >> 2, lk = (tid & 3) * 8;   // tile loader coords (A and B)

  v8f c0 = {}, c1 = {};

  const _Float16* aseg = A + (size_t)(bm + lr) * K + lk;
  const _Float16* bseg = BT + (size_t)(bn + lr) * K + lk;

  for (int kt = 0; kt < K; kt += 32) {
    // Stage A 64x32 and B^T 64x32 tiles: one async 16B DMA each per thread.
    copy16_g2l(aseg + kt, &As[lr][lk]);
    copy16_g2l(bseg + kt, &BsT[lr][lk]);
    if (kt + 32 < K) {
      __builtin_prefetch(aseg + kt + 32, 0, 1);
      __builtin_prefetch(bseg + kt + 32, 0, 1);
    }
    wait_async();
    __syncthreads();

    // A fragment (contiguous K run: lane<16 -> K{0..15}, else K{16..31}).
    const int arow = wm + l15;
    v8h alo = *(const v8h*)&As[arow][halfsel * 16];
    v8h ahi = *(const v8h*)&As[arow][halfsel * 16 + 8];
    v16h a = FULL_SHUF(alo, ahi);
    // B fragments: col = n, contiguous K run selected by half.
    const int kb = halfsel * 16;
    v16h b0 = FULL_SHUF(*(const v8h*)&BsT[wn + l15][kb],
                        *(const v8h*)&BsT[wn + l15][kb + 8]);
    v16h b1 = FULL_SHUF(*(const v8h*)&BsT[wn + 16 + l15][kb],
                        *(const v8h*)&BsT[wn + 16 + l15][kb + 8]);
    c0 = wmma16(a, b0, c0);
    c1 = wmma16(a, b1, c1);
    __syncthreads();
  }

  // Epilogue: C layout -> element i is row (i + 8*half), col = lane&15.
#pragma unroll
  for (int i = 0; i < 8; ++i) {
    const int row = bm + wm + i + halfsel * 8;
    const int col0 = bn + wn + l15;
    const int col1 = col0 + 16;
    float v0 = c0[i] + bias[col0];
    float v1 = c1[i] + bias[col1];
    if (resid) {
      v0 += resid[(size_t)row * N + col0];
      v1 += resid[(size_t)row * N + col1];
    }
    if (act == 1) {  // exact GeLU
      v0 = 0.5f * v0 * (1.0f + erff(v0 * 0.70710678118654752f));
      v1 = 0.5f * v1 * (1.0f + erff(v1 * 0.70710678118654752f));
    }
    if (outF32) {
      outF32[(size_t)row * N + col0] = v0;
      outF32[(size_t)row * N + col1] = v1;
    }
    if (outF16) {
      outF16[(size_t)row * N + col0] = (_Float16)v0;
      outF16[(size_t)row * N + col1] = (_Float16)v1;
    }
    if (outVT) {  // per-head transposed: vt[((b*12+h)*64+d)*4096 + s]
      const int bidx = row >> 12, sloc = row & 4095;
      const int h0 = col0 >> 6, d0 = col0 & 63;
      const int h1 = col1 >> 6, d1 = col1 & 63;
      outVT[(((size_t)bidx * 12 + h0) * 64 + d0) * 4096 + sloc] = (_Float16)v0;
      outVT[(((size_t)bidx * 12 + h1) * 64 + d1) * 4096 + sloc] = (_Float16)v1;
    }
  }
}

// ---------------------------------------------------------------------------
// BigBird attention: one block per (b,h,query-block). 4 waves, each owns 16
// query rows; flash-style online softmax over the 8 attended 64-key blocks.
// K tiles and V^T tiles are both contiguous async DMAs into LDS.
// ---------------------------------------------------------------------------
__global__ void __launch_bounds__(128)
k_attention(const _Float16* __restrict__ Q, const _Float16* __restrict__ Kk,
            const _Float16* __restrict__ Vt, const int* __restrict__ idx,
            _Float16* __restrict__ Out) {
  __shared__ _Float16 Qs[64][72];      // [query tok][dim]
  __shared__ _Float16 Ks[64][72];      // [key tok][dim]  == BsT for scores
  __shared__ _Float16 VsT[64][72];     // [dim][key tok]  == BsT for PV
  __shared__ _Float16 Ps[4][16][72];   // wave-private probs (C->A relayout)

  const int n = blockIdx.x;  // query block (0..63)
  const int h = blockIdx.y;  // head (0..11)
  const int b = blockIdx.z;  // batch (0..3)
  const int tid = threadIdx.x;
  const int wave = tid >> 5;
  const int lane = tid & 31;
  const int l15 = lane & 15;
  const int halfsel = lane >> 4;

  const size_t hoff = (size_t)h * 64;
  const size_t qrow0 = (size_t)(b * 4096 + n * 64);
  const size_t vplane = ((size_t)(b * 12 + h) * 64) * 4096;  // Vt[b][h][.][.]

  // Load Q block (64x64 f16).
#pragma unroll
  for (int it = 0; it < 4; ++it) {
    int L = it * 128 + tid;
    int tok = L >> 3, d0 = (L & 7) * 8;
    copy16_g2l(Q + (qrow0 + tok) * 768 + hoff + d0, &Qs[tok][d0]);
  }
  wait_async();
  __syncthreads();

  float rmax[8], rsum[8];
  v8f o[4] = {};
#pragma unroll
  for (int i = 0; i < 8; ++i) { rmax[i] = -3.0e38f; rsum[i] = 0.0f; }

  for (int m = 0; m < 8; ++m) {
    const int mblk = idx[n * 8 + m];
    const size_t krow0 = (size_t)(b * 4096 + mblk * 64);
    __syncthreads();  // previous iteration's K/V reads complete
#pragma unroll
    for (int it = 0; it < 4; ++it) {
      int L = it * 128 + tid;
      int r = L >> 3, c0 = (L & 7) * 8;
      copy16_g2l(Kk + (krow0 + r) * 768 + hoff + c0, &Ks[r][c0]);            // [tok][dim]
      copy16_g2l(Vt + vplane + (size_t)r * 4096 + mblk * 64 + c0, &VsT[r][c0]);  // [dim][tok]
    }
    wait_async();
    __syncthreads();

    // scores strip: 16 query rows x 64 keys, K(=head dim)=64 in two steps
    v8f s[4] = {};
    const int arow = wave * 16 + l15;
#pragma unroll
    for (int ks = 0; ks < 64; ks += 32) {
      v8h alo = *(const v8h*)&Qs[arow][ks + halfsel * 16];
      v8h ahi = *(const v8h*)&Qs[arow][ks + halfsel * 16 + 8];
      v16h a = FULL_SHUF(alo, ahi);
      const int kb = ks + halfsel * 16;
#pragma unroll
      for (int g = 0; g < 4; ++g) {
        v16h bb = FULL_SHUF(*(const v8h*)&Ks[g * 16 + l15][kb],
                            *(const v8h*)&Ks[g * 16 + l15][kb + 8]);
        s[g] = wmma16(a, bb, s[g]);
      }
    }

    // online softmax: acc element i <-> row (i + 8*half); reduce across the
    // 16 lanes of each half (those are the 16 columns of one 16-wide group).
#pragma unroll
    for (int i = 0; i < 8; ++i) {
      float xs[4];
#pragma unroll
      for (int g = 0; g < 4; ++g) xs[g] = s[g][i] * 0.125f;  // 1/sqrt(64)
      float mv = fmaxf(fmaxf(xs[0], xs[1]), fmaxf(xs[2], xs[3]));
#pragma unroll
      for (int off = 1; off < 16; off <<= 1)
        mv = fmaxf(mv, __shfl_xor(mv, off, 16));
      const float nm = fmaxf(rmax[i], mv);
      const float corr = __expf(rmax[i] - nm);
      float psum = 0.0f;
      const int pr = i + halfsel * 8;
#pragma unroll
      for (int g = 0; g < 4; ++g) {
        float p = __expf(xs[g] - nm);
        psum += p;
        Ps[wave][pr][g * 16 + l15] = (_Float16)p;  // C-layout -> row-major
      }
#pragma unroll
      for (int off = 1; off < 16; off <<= 1)
        psum += __shfl_xor(psum, off, 16);
      rsum[i] = rsum[i] * corr + psum;
      rmax[i] = nm;
#pragma unroll
      for (int g = 0; g < 4; ++g) o[g][i] *= corr;
    }

    // PV: A = probs (16x64 keys), B = V^T[dim][key]; accumulate into o[].
#pragma unroll
    for (int ks = 0; ks < 64; ks += 32) {
      v8h plo = *(const v8h*)&Ps[wave][l15][ks + halfsel * 16];
      v8h phi = *(const v8h*)&Ps[wave][l15][ks + halfsel * 16 + 8];
      v16h a = FULL_SHUF(plo, phi);
      const int kb = ks + halfsel * 16;
#pragma unroll
      for (int g = 0; g < 4; ++g) {
        v16h bb = FULL_SHUF(*(const v8h*)&VsT[g * 16 + l15][kb],
                            *(const v8h*)&VsT[g * 16 + l15][kb + 8]);
        o[g] = wmma16(a, bb, o[g]);
      }
    }
  }

  // Normalize by row sums and store f16 output at [b, s, h*64 + d].
#pragma unroll
  for (int i = 0; i < 8; ++i) {
    const int r = wave * 16 + i + halfsel * 8;
    const float inv = 1.0f / rsum[i];
    _Float16* orow = Out + (qrow0 + r) * 768 + hoff;
#pragma unroll
    for (int g = 0; g < 4; ++g)
      orow[g * 16 + l15] = (_Float16)(o[g][i] * inv);
  }
}

// ---------------------------------------------------------------------------
// Host orchestration
// ---------------------------------------------------------------------------
extern "C" void kernel_launch(void* const* d_in, const int* in_sizes, int n_in,
                              void* d_out, int out_size, void* d_ws, size_t ws_size,
                              hipStream_t stream) {
  (void)in_sizes; (void)n_in; (void)out_size; (void)ws_size;
  const float* x    = (const float*)d_in[0];
  const float* ln1w = (const float*)d_in[1];
  const float* ln1b = (const float*)d_in[2];
  const float* Wq   = (const float*)d_in[3];
  const float* bq   = (const float*)d_in[4];
  const float* Wk   = (const float*)d_in[5];
  const float* bk   = (const float*)d_in[6];
  const float* Wv   = (const float*)d_in[7];
  const float* bv   = (const float*)d_in[8];
  const float* Wo   = (const float*)d_in[9];
  const float* bo   = (const float*)d_in[10];
  const float* ln2w = (const float*)d_in[11];
  const float* ln2b = (const float*)d_in[12];
  const float* W1   = (const float*)d_in[13];
  const float* b1   = (const float*)d_in[14];
  const float* W2   = (const float*)d_in[15];
  const float* b2   = (const float*)d_in[16];
  float* out = (float*)d_out;

  const int M = 16384, D = 768, F = 3072;

  char* ws = (char*)d_ws;
  size_t off = 0;
  auto alloc = [&](size_t bytes) -> void* {
    size_t o = off;
    off = (off + bytes + 255) & ~(size_t)255;
    return (void*)(ws + o);
  };

  _Float16* WqT = (_Float16*)alloc((size_t)D * D * 2);  // [N][K] f16
  _Float16* WkT = (_Float16*)alloc((size_t)D * D * 2);
  _Float16* WvT = (_Float16*)alloc((size_t)D * D * 2);
  _Float16* WoT = (_Float16*)alloc((size_t)D * D * 2);
  _Float16* W1T = (_Float16*)alloc((size_t)D * F * 2);  // [F][D]
  _Float16* W2T = (_Float16*)alloc((size_t)F * D * 2);  // [D][F]
  _Float16* h16 = (_Float16*)alloc((size_t)M * D * 2);  // LN1 out, reused for LN2 out
  _Float16* q16 = (_Float16*)alloc((size_t)M * D * 2);
  _Float16* k16 = (_Float16*)alloc((size_t)M * D * 2);
  _Float16* vT  = (_Float16*)alloc((size_t)M * D * 2);  // [b][h][d][s]
  _Float16* a16 = (_Float16*)alloc((size_t)M * D * 2);  // attention out
  float*    x1  = (float*)alloc((size_t)M * D * 4);     // post-attn residual
  _Float16* f1  = (_Float16*)alloc((size_t)M * F * 2);  // FFN hidden
  int*      idx = (int*)alloc(64 * 8 * sizeof(int));

  auto cvtT = [&](const float* s, _Float16* dst, int N, int K) {
    int n = N * K;
    k_cvt_t<<<(n + 255) / 256, 256, 0, stream>>>(s, dst, N, K);
  };
  cvtT(Wq, WqT, D, D);
  cvtT(Wk, WkT, D, D);
  cvtT(Wv, WvT, D, D);
  cvtT(Wo, WoT, D, D);
  cvtT(W1, W1T, F, D);
  cvtT(W2, W2T, D, F);

  k_build_idx<<<1, 64, 0, stream>>>(idx);

  // LN1
  k_layernorm<<<M, 256, 0, stream>>>(x, ln1w, ln1b, h16);

  dim3 gD(D / 64, M / 64);   // (12, 256)
  dim3 gF(F / 64, M / 64);   // (48, 256)

  // Q, K, V projections (f16 outputs; V also written head-transposed)
  k_gemm<<<gD, 256, 0, stream>>>(h16, WqT, bq, nullptr, nullptr, q16, nullptr,
                                 M, D, D, 0);
  k_gemm<<<gD, 256, 0, stream>>>(h16, WkT, bk, nullptr, nullptr, k16, nullptr,
                                 M, D, D, 0);
  k_gemm<<<gD, 256, 0, stream>>>(h16, WvT, bv, nullptr, nullptr, nullptr, vT,
                                 M, D, D, 0);

  // Block-sparse attention: grid (query blocks, heads, batch)
  k_attention<<<dim3(64, 12, 4), 128, 0, stream>>>(q16, k16, vT, idx, a16);

  // O projection + residual(x) -> x1 (f32)
  k_gemm<<<gD, 256, 0, stream>>>(a16, WoT, bo, x, x1, nullptr, nullptr,
                                 M, D, D, 0);

  // LN2 -> h16 (f16)
  k_layernorm<<<M, 256, 0, stream>>>(x1, ln2w, ln2b, h16);

  // FFN1 + exact GeLU -> f1 (f16)
  k_gemm<<<gF, 256, 0, stream>>>(h16, W1T, b1, nullptr, nullptr, f1, nullptr,
                                 M, F, D, 1);

  // FFN2 + residual(x1) -> out (f32)
  k_gemm<<<gD, 256, 0, stream>>>(f1, W2T, b2, x1, out, nullptr, nullptr,
                                 M, D, F, 0);
}
